// AttnAugmentation2d_29566554865758
// MI455X (gfx1250) — matrix operand order repaired and checked by reading, not compiled, not run on hardware
//
#include <hip/hip_runtime.h>

// ---------------------------------------------------------------------------
// Attention-augmented 2D conv attention for MI455X (gfx1250, wave32, WMMA).
// B=8, heads=8, H=W=32 (L=1024), dk_head=dv_head=32.
// Flash-style streaming softmax; K/V converted f32->bf16 ONCE per workgroup
// (v_cvt_pk_bf16_f32) into double-buffered LDS in WMMA-fragment-ready
// packed-pair layout; fragments are whole-vector bit_casts of ds_load_b32
// results so no 16-bit re-splicing VALU appears in the hot loop.
// ---------------------------------------------------------------------------

typedef __attribute__((ext_vector_type(16))) __bf16    v16bf;
typedef __attribute__((ext_vector_type(8)))  float     v8f;
typedef __attribute__((ext_vector_type(8)))  unsigned  v8u;
typedef __attribute__((ext_vector_type(2)))  __bf16    bf2;

// Pack two f32 into one dword of bf16 pair (low half = first arg), RNE.
__device__ __forceinline__ unsigned pack2(float lo, float hi) {
#if __has_builtin(__builtin_amdgcn_cvt_pk_bf16_f32)
  bf2 r = __builtin_amdgcn_cvt_pk_bf16_f32(lo, hi);
  return __builtin_bit_cast(unsigned, r);
#else
  unsigned r;
  asm("v_cvt_pk_bf16_f32 %0, %1, %2" : "=v"(r) : "v"(lo), "v"(hi));
  return r;
#endif
}

#define NW 4  // waves per workgroup

__global__ __launch_bounds__(128) void attn_aug2d_wmma(
    const float* __restrict__ x,      // [8, 768, 32, 32]
    const float* __restrict__ relw,   // [63, 32]
    const float* __restrict__ relh,   // [63, 32]
    float* __restrict__ out)          // [8, 256, 32, 32]
{
  constexpr int L  = 1024;   // H*W
  constexpr int Dh = 32;     // per-head depth
  constexpr int CH = 768;
  const float scale = 0.17677669529663687f;  // 1/sqrt(32)

  // Workgroup-shared double-buffered K/V tiles, packed bf16 pairs:
  //   s_k[buf][fp][key32]  = { K[feat 2fp][kb+key32], K[feat 2fp+1][...] }
  //   s_v[buf][kp][feat]   = { V[key kb+2kp][feat],   V[key kb+2kp+1][feat] }
  __shared__ unsigned s_k[2][16][36];
  __shared__ unsigned s_v[2][16][36];
  // Per-wave private regions.
  __shared__ float  s_rw[NW][16][66];   // RW rel table: 16 q rows x 63(+pad)
  __shared__ float  s_rh[NW][16][66];   // RH rel table
  __shared__ __bf16 s_p [NW][16][36];   // P staging (C-layout -> A-layout)

  const int tid  = threadIdx.x;
  const int wv   = tid >> 5;
  const int lane = tid & 31;
  const int lh   = lane >> 4;   // lane half (0: lanes 0-15, 1: lanes 16-31)
  const int ln   = lane & 15;

  const int bid   = blockIdx.x;          // 1024 blocks
  const int bh    = bid >> 4;            // b*8 + head
  const int b     = bh >> 3;
  const int head  = bh & 7;
  const int qbase = (bid & 15) * 64 + wv * 16;   // first query row of wave

  // x memory: [b][c][l]; per-head Q/K/V are [feature d][position l].
  const float* qp = x + ((size_t)b * CH + head * Dh) * L;
  const float* kp = qp + (size_t)256 * L;
  const float* vp = qp + (size_t)512 * L;

  // Cooperative fill of one 32-key K/V block into LDS buffer `buf`.
  auto fill = [&](int buf, int kb) {
    const int rowi = tid >> 3;          // 0..15
    const int c4   = (tid & 7) * 4;     // 0,4,...,28
    {
      const float* r0 = kp + (size_t)(2 * rowi) * L + kb + c4;
      float4 a = *(const float4*)r0;
      float4 c = *(const float4*)(r0 + L);
      uint4 o;
      o.x = pack2(a.x, c.x); o.y = pack2(a.y, c.y);
      o.z = pack2(a.z, c.z); o.w = pack2(a.w, c.w);
      *(uint4*)&s_k[buf][rowi][c4] = o;
      if (kb + 32 < L) __builtin_prefetch(r0 + 32, 0, 3);  // warm next block
    }
    {
      uint4 o;
      #pragma unroll
      for (int i = 0; i < 4; ++i) {
        float2 f = *(const float2*)(vp + (size_t)(c4 + i) * L + kb + 2 * rowi);
        ((unsigned*)&o)[i] = pack2(f.x, f.y);
      }
      *(uint4*)&s_v[buf][rowi][c4] = o;
    }
  };

  // ---- Q tile (16 rows x 32 feats) into bf16 A-layout, pre-scaled --------
  // A operand in this code's contiguous k order (B matches): lanes0-15 M=lane: K=0..15;
  // lanes16-31: K=16..31. Packed pair per VGPR: low half = lower K.
  v8u qu;
  #pragma unroll
  for (int v = 0; v < 8; ++v) {
    const int kk0 = lh * 16 + 2 * v;
    float f0 = qp[(size_t)kk0 * L + (qbase + ln)] * scale;
    float f1 = qp[(size_t)(kk0 + 1) * L + (qbase + ln)] * scale;
    qu[v] = pack2(f0, f1);
  }
  const v16bf qa = __builtin_bit_cast(v16bf, qu);

  fill(0, 0);   // stage first K/V block while we build the rel tables

  // ---- Relative-logit tables via WMMA: RW = Q x rel_w^T, RH = Q x rel_h^T --
  // B operand in the same contiguous k order as qa: N = lane%16; lanes0-15 K=0..15, lanes16-31
  // K=16..31, VGPR v packs K = half*16 + {2v, 2v+1}.
  #pragma unroll
  for (int which = 0; which < 2; ++which) {
    const float* rp = which ? relh : relw;
    float (*dst)[66] = which ? s_rh[wv] : s_rw[wv];
    #pragma unroll
    for (int t = 0; t < 4; ++t) {
      const int n  = t * 16 + ln;          // rel index m (63 valid, pad to 64)
      const int nc = n < 63 ? n : 62;      // clamped: no EXEC-divergent load
      const float msk = n < 63 ? 1.f : 0.f;
      v8u bu;
      #pragma unroll
      for (int v = 0; v < 8; ++v) {
        const int kk = lh * 16 + 2 * v;
        float2 f2 = *(const float2*)(rp + nc * 32 + kk);
        bu[v] = pack2(f2.x * msk, f2.y * msk);
      }
      v16bf bb = __builtin_bit_cast(v16bf, bu);
      v8f c = {};
      c = __builtin_amdgcn_wmma_f32_16x16x32_bf16(false, qa, false, bb,
                                                  (short)0, c, false, false);
      #pragma unroll
      for (int r = 0; r < 8; ++r)
        dst[r + 8 * lh][t * 16 + ln] = c[r];
    }
  }
  __syncthreads();   // first K/V block staged; rel tables visible

  // Loop-invariant rel-table element offsets (flattened rows of 66 floats).
  const float* rwb = &s_rw[wv][0][0];
  const float* rhb = &s_rh[wv][0][0];
  int rw_off[8], rh_off[8];
  #pragma unroll
  for (int r = 0; r < 8; ++r) {
    const int Mr = r + 8 * lh;
    const int ql = qbase + Mr;
    rw_off[r] = Mr * 66 + (ln - (ql & 31) + 31);
    rh_off[r] = Mr * 66 + (31 - (ql >> 5));     // + pr each iteration
  }

  // ---- Flash-attention main loop over 32 key blocks of 32 -----------------
  v8f O0 = {}, O1 = {};                 // O tile 16x32 f32 (two C tiles)
  float mrow[8], lrow[8];
  #pragma unroll
  for (int r = 0; r < 8; ++r) { mrow[r] = -1e30f; lrow[r] = 0.f; }

  #pragma unroll 2
  for (int it = 0; it < 32; ++it) {
    const int kb  = it << 5;
    const int cur = it & 1;
    if (it + 1 < 32) fill(cur ^ 1, kb + 32);   // overlap next block's loads

    // K B-tiles straight out of LDS (keys kb..kb+15 / kb+16..kb+31).
    v8u ku0, ku1;
    #pragma unroll
    for (int v = 0; v < 8; ++v) {
      ku0[v] = s_k[cur][lh * 8 + v][ln];
      ku1[v] = s_k[cur][lh * 8 + v][16 + ln];
    }
    v16bf kt0 = __builtin_bit_cast(v16bf, ku0);
    v16bf kt1 = __builtin_bit_cast(v16bf, ku1);
    v8f S0 = {}, S1 = {};
    S0 = __builtin_amdgcn_wmma_f32_16x16x32_bf16(false, qa, false, kt0,
                                                 (short)0, S0, false, false);
    S1 = __builtin_amdgcn_wmma_f32_16x16x32_bf16(false, qa, false, kt1,
                                                 (short)0, S1, false, false);

    // Relative logits. Keys kb..kb+31 all sit in image row pr = kb>>5.
    const int pr = kb >> 5;
    #pragma unroll
    for (int r = 0; r < 8; ++r) {
      const float rh_add = rhb[rh_off[r] + pr];
      S0[r] += rwb[rw_off[r]]      + rh_add;    // key col ln
      S1[r] += rwb[rw_off[r] + 16] + rh_add;    // key col 16+ln
    }

    // Online softmax per query row (row lives in one 16-lane half).
    #pragma unroll
    for (int r = 0; r < 8; ++r) {
      float mx = fmaxf(S0[r], S1[r]);
      #pragma unroll
      for (int m = 1; m < 16; m <<= 1) mx = fmaxf(mx, __shfl_xor(mx, m, 32));
      const float mnew  = fmaxf(mrow[r], mx);
      const float alpha = __expf(mrow[r] - mnew);
      mrow[r] = mnew;
      const float p0 = __expf(S0[r] - mnew);
      const float p1 = __expf(S1[r] - mnew);
      float rs = p0 + p1;
      #pragma unroll
      for (int m = 1; m < 16; m <<= 1) rs += __shfl_xor(rs, m, 32);
      lrow[r] = lrow[r] * alpha + rs;
      O0[r] *= alpha;
      O1[r] *= alpha;
      const int Mr = r + 8 * lh;
      bf2 pb = __builtin_bit_cast(bf2, pack2(p0, p1));   // one v_cvt_pk
      s_p[wv][Mr][ln]      = pb.x;
      s_p[wv][Mr][16 + ln] = pb.y;
    }
    asm volatile("s_wait_dscnt 0" ::: "memory");  // wave-local P RAW fence

    // P back out of LDS in A-layout (adjacent-K pair -> one b32 load).
    v8u pu;
    #pragma unroll
    for (int v = 0; v < 8; ++v) {
      const int kk0 = lh * 16 + 2 * v;
      pu[v] = *(const unsigned*)&s_p[wv][ln][kk0];
    }
    v16bf pa = __builtin_bit_cast(v16bf, pu);

    // V B-tiles from LDS (key pair kp = lh*8+v, feature col = tile*16+ln).
    v8u vu0, vu1;
    #pragma unroll
    for (int v = 0; v < 8; ++v) {
      vu0[v] = s_v[cur][lh * 8 + v][ln];
      vu1[v] = s_v[cur][lh * 8 + v][16 + ln];
    }
    v16bf vb0 = __builtin_bit_cast(v16bf, vu0);
    v16bf vb1 = __builtin_bit_cast(v16bf, vu1);
    O0 = __builtin_amdgcn_wmma_f32_16x16x32_bf16(false, pa, false, vb0,
                                                 (short)0, O0, false, false);
    O1 = __builtin_amdgcn_wmma_f32_16x16x32_bf16(false, pa, false, vb1,
                                                 (short)0, O1, false, false);

    __syncthreads();   // next buffer staged; current buffer free for reuse
  }

  // ---- normalize and store NCHW: out[b][head*32+d][l] ---------------------
  float* ob = out + ((size_t)b * 256 + head * 32) * L;
  #pragma unroll
  for (int r = 0; r < 8; ++r) {
    const int Mr = r + 8 * lh;
    const float inv = 1.0f / lrow[r];
    ob[(size_t)ln * L        + qbase + Mr] = O0[r] * inv;   // feats 0..15
    ob[(size_t)(16 + ln) * L + qbase + Mr] = O1[r] * inv;   // feats 16..31
  }
}

extern "C" void kernel_launch(void* const* d_in, const int* in_sizes, int n_in,
                              void* d_out, int out_size, void* d_ws, size_t ws_size,
                              hipStream_t stream) {
  (void)in_sizes; (void)n_in; (void)out_size; (void)d_ws; (void)ws_size;
  const float* x    = (const float*)d_in[0];
  const float* relw = (const float*)d_in[1];
  const float* relh = (const float*)d_in[2];
  float* out = (float*)d_out;
  // 64 (b,head) pairs x 16 query blocks of 64 rows; 128 threads = 4 wave32.
  hipLaunchKernelGGL(attn_aug2d_wmma, dim3(1024), dim3(128), 0, stream,
                     x, relw, relh, out);
}
